// Seq2SeqAttention_60060822667549
// MI455X (gfx1250) — matrix operand order repaired and checked by reading, not compiled, not run on hardware
//
#include <hip/hip_runtime.h>
#include <hip/hip_bf16.h>

// ---------------------------------------------------------------------------
// Seq2Seq (biLSTM encoder + attention decoder) for gfx1250 / MI455X.
// All GEMM work runs on v_wmma_f32_16x16x32_f16 (f16 in, f32 accumulate).
// B==16 matches the WMMA M tile exactly, so every recurrent step is a row of
// 16x16 WMMA tiles. Generic GEMM waves own 4 N-tiles sharing one A-tile load
// per K chunk (2.5 b128 loads per WMMA). Weights converted to f16 in d_ws.
// ---------------------------------------------------------------------------

typedef _Float16 h16v __attribute__((ext_vector_type(16)));
typedef _Float16 h8v  __attribute__((ext_vector_type(8)));
typedef float    f8v  __attribute__((ext_vector_type(8)));

#define DEVINL __device__ __forceinline__

DEVINL float sigmoidf_(float x) { return 1.0f / (1.0f + expf(-x)); }

// A-matrix 16x32 f16 tile (ISA 7.12.2): lane l<16 -> row l, K = {0..7,16..23};
// lane l+16 -> row l, K = {8..15,24..31}. Two 16B loads per lane.
DEVINL h16v load_a_tile(const _Float16* __restrict__ A, int lda, int l, int hi, int k0) {
  const _Float16* p = A + (size_t)l * lda + k0 + hi * 8;
  union { h8v v[2]; h16v w; } u;
  u.v[0] = *(const h8v*)(p);
  u.v[1] = *(const h8v*)(p + 16);
  return u.w;
}

// B-matrix 32x16 f16 tile: B column n = row `nrow` of W (we compute x @ W^T).
// lane l<16 -> col, K = {0..7,16..23} ; lane l+16 -> col, K = {8..15,24..31}.
DEVINL h16v load_b_tile(const _Float16* __restrict__ W, int ldb, size_t nrow, int hi, int k0) {
  const _Float16* p = W + nrow * (size_t)ldb + k0 + hi * 8;
  union { h8v v[2]; h16v w; } u;
  u.v[0] = *(const h8v*)(p);
  u.v[1] = *(const h8v*)(p + 16);
  return u.w;
}

DEVINL f8v wmma16(h16v a, h16v b, f8v c) {
  return __builtin_amdgcn_wmma_f32_16x16x32_f16(false, a, false, b, (short)0, c, false, false);
}

// ---------------------------------------------------------------------------
// Generic C = act(A @ W^T + bias); A: MxK f16 (lda), W: NxK f16 (ldb=K-major).
// One wave per 16x64 output slab (4 WMMA accumulators sharing the A tile);
// 4 waves / block. Optional f32 and f16 outputs. Ntiles must be %4 == 0.
// ---------------------------------------------------------------------------
__global__ void gemm_wmma_kernel(const _Float16* __restrict__ A, int lda,
                                 const _Float16* __restrict__ W, int ldb,
                                 const float* __restrict__ bias,
                                 float* __restrict__ C32, int ldc32,
                                 _Float16* __restrict__ C16, int ldc16,
                                 int Mtiles, int Nquads, int K, int act) {
  int wave = (int)((blockIdx.x * blockDim.x + threadIdx.x) >> 5);
  if (wave >= Mtiles * Nquads) return;              // wave-uniform: EXEC stays all-1s
  int nq = wave % Nquads, mt = wave / Nquads;
  int lane = threadIdx.x & 31, l = lane & 15, hi = lane >> 4;
  int m0 = mt * 16, n0 = nq * 64;
  const _Float16* Arow = A + (size_t)m0 * lda;
  f8v acc[4] = {};
  for (int k0 = 0; k0 < K; k0 += 32) {
    h16v a  = load_a_tile(Arow, lda, l, hi, k0);
    h16v b0 = load_b_tile(W, ldb, (size_t)(n0 + 0  + l), hi, k0);
    h16v b1 = load_b_tile(W, ldb, (size_t)(n0 + 16 + l), hi, k0);
    h16v b2 = load_b_tile(W, ldb, (size_t)(n0 + 32 + l), hi, k0);
    h16v b3 = load_b_tile(W, ldb, (size_t)(n0 + 48 + l), hi, k0);
    acc[0] = wmma16(a, b0, acc[0]);
    acc[1] = wmma16(a, b1, acc[1]);
    acc[2] = wmma16(a, b2, acc[2]);
    acc[3] = wmma16(a, b3, acc[3]);
  }
#pragma unroll
  for (int j = 0; j < 4; ++j) {
    int n = n0 + j * 16 + l;
    float bval = bias ? bias[n] : 0.0f;
#pragma unroll
    for (int g = 0; g < 8; ++g) {
      int m = m0 + g + hi * 8;
      float v = acc[j][g] + bval;
      if (act) v = tanhf(v);
      if (C32) C32[(size_t)m * ldc32 + n] = v;
      if (C16) C16[(size_t)m * ldc16 + n] = (_Float16)v;
    }
  }
}

// ---------------------------------------------------------------------------
// Fused LSTM cell step: gates = Gx + h_prev @ W_hh^T ; nonlinearity ; c,h
// One wave owns a 16-wide hidden slice -> 4 WMMA accumulators (i,f,g,o)
// sharing one A-tile load per K chunk. Gx already contains x@W_ih^T + biases.
// ---------------------------------------------------------------------------
__global__ void lstm_step_kernel(const _Float16* __restrict__ hprev, int h_lda,
                                 const _Float16* __restrict__ Whh,   // (4*HD, K)
                                 const float* __restrict__ Gx,       // (16, 4*HD)
                                 float* __restrict__ c, int c_ld,
                                 float* __restrict__ h32, long long h32_rs,   // nullable
                                 _Float16* __restrict__ h16a, int h16a_ld,
                                 _Float16* __restrict__ h16b, int h16b_ld,    // nullable
                                 int HD, int K) {
  int wave = (int)((blockIdx.x * blockDim.x + threadIdx.x) >> 5);
  int lane = threadIdx.x & 31, l = lane & 15, hi = lane >> 4;
  int n0 = wave * 16;
  f8v ai = {}, af = {}, ag = {}, ao = {};
  for (int k0 = 0; k0 < K; k0 += 32) {
    h16v a  = load_a_tile(hprev, h_lda, l, hi, k0);
    h16v bi = load_b_tile(Whh, K, (size_t)(0 * HD + n0 + l), hi, k0);
    h16v bf = load_b_tile(Whh, K, (size_t)(1 * HD + n0 + l), hi, k0);
    h16v bg = load_b_tile(Whh, K, (size_t)(2 * HD + n0 + l), hi, k0);
    h16v bo = load_b_tile(Whh, K, (size_t)(3 * HD + n0 + l), hi, k0);
    ai = wmma16(a, bi, ai);
    af = wmma16(a, bf, af);
    ag = wmma16(a, bg, ag);
    ao = wmma16(a, bo, ao);
  }
  int n = n0 + l;
#pragma unroll
  for (int g = 0; g < 8; ++g) {
    int m = g + hi * 8;                       // batch row (B == 16 == M tile)
    const float* gx = Gx + (size_t)m * 4 * HD;
    float iv = sigmoidf_(ai[g] + gx[0 * HD + n]);
    float fv = sigmoidf_(af[g] + gx[1 * HD + n]);
    float gv = tanhf    (ag[g] + gx[2 * HD + n]);
    float ov = sigmoidf_(ao[g] + gx[3 * HD + n]);
    size_t ci = (size_t)m * c_ld + n;
    float cn = fv * c[ci] + iv * gv;          // each (m,n) owned by one lane: in-place OK
    c[ci] = cn;
    float hv = ov * tanhf(cn);
    if (h32) h32[(long long)m * h32_rs + n] = hv;
    h16a[(size_t)m * h16a_ld + n] = (_Float16)hv;
    if (h16b) h16b[(size_t)m * h16b_ld + n] = (_Float16)hv;
  }
}

// ---------------------------------------------------------------------------
// Attention for one decoder step. Block b: scores over 256 src positions,
// softmax, weighted context -> f16 into concat buffer (cols 0..1023, ld 2048).
// ---------------------------------------------------------------------------
__global__ void attn_kernel(const float* __restrict__ ctx,      // (16,256,1024)
                            const float* __restrict__ target,   // (16,1024)
                            _Float16* __restrict__ wctx) {      // ld 2048
  __shared__ float tg[1024];
  __shared__ float sc[256];
  __shared__ float red[256];
  int b = blockIdx.x;
  int t = threadIdx.x;                                   // 256 threads
  for (int h = t; h < 1024; h += 256) tg[h] = target[b * 1024 + h];
  __syncthreads();
  const float* crow = ctx + ((size_t)b * 256 + t) * 1024;
  float dot = 0.0f;
  for (int h = 0; h < 1024; ++h) dot += crow[h] * tg[h];
  red[t] = dot;
  __syncthreads();
  for (int off = 128; off > 0; off >>= 1) {
    if (t < off) red[t] = fmaxf(red[t], red[t + off]);
    __syncthreads();
  }
  float mx = red[0];
  __syncthreads();
  float e = expf(dot - mx);
  red[t] = e;
  __syncthreads();
  for (int off = 128; off > 0; off >>= 1) {
    if (t < off) red[t] += red[t + off];
    __syncthreads();
  }
  float inv = 1.0f / red[0];
  __syncthreads();
  sc[t] = e * inv;
  __syncthreads();
  const float* cbase = ctx + (size_t)b * 256 * 1024;
  for (int h = t; h < 1024; h += 256) {
    float acc = 0.0f;
    for (int s = 0; s < 256; ++s) acc += sc[s] * cbase[(size_t)s * 1024 + h];
    wctx[(size_t)b * 2048 + h] = (_Float16)acc;
  }
}

// ---------------------------------------------------------------------------
// Small utility kernels
// ---------------------------------------------------------------------------
__global__ void f32_to_f16_kernel(const float* __restrict__ s, _Float16* __restrict__ d, size_t n) {
  for (size_t i = blockIdx.x * (size_t)blockDim.x + threadIdx.x; i < n;
       i += (size_t)gridDim.x * blockDim.x)
    d[i] = (_Float16)s[i];
}
__global__ void gather_emb_f16_kernel(const int* __restrict__ idx, const float* __restrict__ emb,
                                      _Float16* __restrict__ out, int E) {
  int r = blockIdx.x;
  const float* src = emb + (size_t)idx[r] * E;
  for (int c = threadIdx.x; c < E; c += blockDim.x)
    out[(size_t)r * E + c] = (_Float16)src[c];
}
__global__ void add_vec_kernel(const float* a, const float* b, float* o, int n) {
  int i = blockIdx.x * blockDim.x + threadIdx.x;
  if (i < n) o[i] = a[i] + b[i];
}
__global__ void zero_f32_kernel(float* p, size_t n) {
  for (size_t i = blockIdx.x * (size_t)blockDim.x + threadIdx.x; i < n;
       i += (size_t)gridDim.x * blockDim.x) p[i] = 0.0f;
}
__global__ void zero_f16_kernel(_Float16* p, size_t n) {
  for (size_t i = blockIdx.x * (size_t)blockDim.x + threadIdx.x; i < n;
       i += (size_t)gridDim.x * blockDim.x) p[i] = (_Float16)0.0f;
}
__global__ void copy2d_f16_kernel(const _Float16* __restrict__ s, int slda,
                                  _Float16* __restrict__ d, int dlda, int cols) {
  int r = blockIdx.x;
  for (int c = threadIdx.x; c < cols; c += blockDim.x)
    d[(size_t)r * dlda + c] = s[(size_t)r * slda + c];
}

// ---------------------------------------------------------------------------
extern "C" void kernel_launch(void* const* d_in, const int* in_sizes, int n_in,
                              void* d_out, int out_size, void* d_ws, size_t ws_size,
                              hipStream_t stream) {
  (void)in_sizes; (void)n_in; (void)out_size; (void)ws_size;
  const int SRC_LEN = 256, TRG_LEN = 64, Bsz = 16;
  const int E = 512, HD = 512, H = 1024, V = 32000;

  const int*   input_src = (const int*)d_in[0];
  const int*   input_trg = (const int*)d_in[1];
  const float* src_emb   = (const float*)d_in[2];
  const float* trg_emb   = (const float*)d_in[3];
  const float* W_ih_f    = (const float*)d_in[4];
  const float* W_hh_f    = (const float*)d_in[5];
  const float* b_ih_f    = (const float*)d_in[6];
  const float* b_hh_f    = (const float*)d_in[7];
  const float* W_ih_b    = (const float*)d_in[8];
  const float* W_hh_b    = (const float*)d_in[9];
  const float* b_ih_b    = (const float*)d_in[10];
  const float* b_hh_b    = (const float*)d_in[11];
  const float* W_e2d     = (const float*)d_in[12];
  const float* b_e2d     = (const float*)d_in[13];
  const float* W_in_dec  = (const float*)d_in[14];
  const float* b_in_dec  = (const float*)d_in[15];
  const float* W_hid_dec = (const float*)d_in[16];
  const float* b_hid_dec = (const float*)d_in[17];
  const float* W_attn_in = (const float*)d_in[18];
  const float* W_attn_out= (const float*)d_in[19];
  const float* W_vocab   = (const float*)d_in[20];
  const float* b_vocab   = (const float*)d_in[21];

  // ---- workspace carve (≈200 MB total) -----------------------------------
  size_t off = 0;
  auto a16 = [&](size_t elems) -> _Float16* {
    _Float16* p = (_Float16*)((char*)d_ws + off);
    off += ((elems * sizeof(_Float16) + 255) / 256) * 256;
    return p;
  };
  auto a32 = [&](size_t elems) -> float* {
    float* p = (float*)((char*)d_ws + off);
    off += ((elems * sizeof(float) + 255) / 256) * 256;
    return p;
  };
  _Float16* w_ih16f   = a16((size_t)4 * HD * E);
  _Float16* w_hh16f   = a16((size_t)4 * HD * HD);
  _Float16* w_ih16b   = a16((size_t)4 * HD * E);
  _Float16* w_hh16b   = a16((size_t)4 * HD * HD);
  _Float16* w_e2d16   = a16((size_t)H * H);
  _Float16* w_indec16 = a16((size_t)4 * H * E);
  _Float16* w_hiddec16= a16((size_t)4 * H * H);
  _Float16* w_attnin16= a16((size_t)H * H);
  _Float16* w_attnout16=a16((size_t)H * 2 * H);
  _Float16* w_vocab16 = a16((size_t)V * H);
  _Float16* srcx16    = a16((size_t)SRC_LEN * Bsz * E);
  _Float16* trgx16    = a16((size_t)TRG_LEN * Bsz * E);
  float*    bias_ef   = a32(4 * HD);
  float*    bias_eb   = a32(4 * HD);
  float*    bias_d    = a32(4 * H);
  float*    enc_gx_f  = a32((size_t)SRC_LEN * Bsz * 4 * HD);
  float*    enc_gx_b  = a32((size_t)SRC_LEN * Bsz * 4 * HD);
  float*    dec_gx    = a32((size_t)TRG_LEN * Bsz * 4 * H);
  float*    ctx       = a32((size_t)Bsz * SRC_LEN * 2 * HD);   // (b, s, 1024)
  float*    c_enc     = a32((size_t)Bsz * 2 * HD);             // [cb | cf] -> decoder c
  _Float16* hfA       = a16((size_t)Bsz * HD);                 // fwd h ping
  _Float16* hfB       = a16((size_t)Bsz * HD);
  _Float16* hbA       = a16((size_t)Bsz * HD);                 // bwd h ping
  _Float16* hbB       = a16((size_t)Bsz * HD);
  _Float16* henc_cat  = a16((size_t)Bsz * 2 * HD);             // [hb | hf]
  _Float16* dec_h0_16 = a16((size_t)Bsz * H);
  _Float16* hy16      = a16((size_t)Bsz * H);
  _Float16* concatA   = a16((size_t)Bsz * 2 * H);              // [wctx | hy]
  float*    target32  = a32((size_t)Bsz * H);
  _Float16* trg_h16   = a16((size_t)TRG_LEN * Bsz * H);

  // ---- weight / input conversion -----------------------------------------
  auto conv = [&](const float* s, _Float16* d, size_t n) {
    f32_to_f16_kernel<<<1024, 256, 0, stream>>>(s, d, n);
  };
  conv(W_ih_f,    w_ih16f,    (size_t)4 * HD * E);
  conv(W_hh_f,    w_hh16f,    (size_t)4 * HD * HD);
  conv(W_ih_b,    w_ih16b,    (size_t)4 * HD * E);
  conv(W_hh_b,    w_hh16b,    (size_t)4 * HD * HD);
  conv(W_e2d,     w_e2d16,    (size_t)H * H);
  conv(W_in_dec,  w_indec16,  (size_t)4 * H * E);
  conv(W_hid_dec, w_hiddec16, (size_t)4 * H * H);
  conv(W_attn_in, w_attnin16, (size_t)H * H);
  conv(W_attn_out,w_attnout16,(size_t)H * 2 * H);
  conv(W_vocab,   w_vocab16,  (size_t)V * H);

  gather_emb_f16_kernel<<<SRC_LEN * Bsz, 256, 0, stream>>>(input_src, src_emb, srcx16, E);
  gather_emb_f16_kernel<<<TRG_LEN * Bsz, 256, 0, stream>>>(input_trg, trg_emb, trgx16, E);

  add_vec_kernel<<<(4 * HD + 255) / 256, 256, 0, stream>>>(b_ih_f, b_hh_f, bias_ef, 4 * HD);
  add_vec_kernel<<<(4 * HD + 255) / 256, 256, 0, stream>>>(b_ih_b, b_hh_b, bias_eb, 4 * HD);
  add_vec_kernel<<<(4 * H + 255) / 256, 256, 0, stream>>>(b_in_dec, b_hid_dec, bias_d, 4 * H);

  zero_f16_kernel<<<16, 256, 0, stream>>>(hfA, (size_t)Bsz * HD);
  zero_f16_kernel<<<16, 256, 0, stream>>>(hbA, (size_t)Bsz * HD);
  zero_f32_kernel<<<16, 256, 0, stream>>>(c_enc, (size_t)Bsz * 2 * HD);

  // Ntiles must be divisible by 4 (each wave owns a 16x64 slab).
  auto gemm = [&](const _Float16* A, int lda, const _Float16* W, int ldb,
                  const float* bias, float* C32, int ldc32, _Float16* C16, int ldc16,
                  int Mtiles, int Ntiles, int K, int act) {
    int waves = Mtiles * (Ntiles / 4);
    gemm_wmma_kernel<<<(waves + 3) / 4, 128, 0, stream>>>(A, lda, W, ldb, bias, C32, ldc32,
                                                          C16, ldc16, Mtiles, Ntiles / 4, K, act);
  };

  // ---- precompute input-gate GEMMs (parallel over all timesteps) ---------
  gemm(srcx16, E, w_ih16f,   E, bias_ef, enc_gx_f, 4 * HD, nullptr, 0, SRC_LEN * Bsz / 16, 4 * HD / 16, E, 0);
  gemm(srcx16, E, w_ih16b,   E, bias_eb, enc_gx_b, 4 * HD, nullptr, 0, SRC_LEN * Bsz / 16, 4 * HD / 16, E, 0);
  gemm(trgx16, E, w_indec16, E, bias_d,  dec_gx,   4 * H,  nullptr, 0, TRG_LEN * Bsz / 16, 4 * H  / 16, E, 0);

  // ---- bidirectional encoder scans (sequential) --------------------------
  const long long ctx_rs = (long long)SRC_LEN * 2 * HD;  // batch-row stride in ctx
  _Float16* fped[2] = { hfA, hfB };
  _Float16* bped[2] = { hbA, hbB };
  for (int t = 0; t < SRC_LEN; ++t) {
    // forward: writes ctx[:, t, 0:512]
    lstm_step_kernel<<<HD / 64, 128, 0, stream>>>(
        fped[t & 1], HD, w_hh16f, enc_gx_f + (size_t)t * Bsz * 4 * HD,
        c_enc + HD /*cf*/, 2 * HD,
        ctx + (size_t)t * 2 * HD, ctx_rs,
        fped[(t + 1) & 1], HD, nullptr, 0, HD, HD);
    // backward: consumes input 255-t, writes ctx[:, 255-t, 512:1024]
    int s = SRC_LEN - 1 - t;
    lstm_step_kernel<<<HD / 64, 128, 0, stream>>>(
        bped[t & 1], HD, w_hh16b, enc_gx_b + (size_t)s * Bsz * 4 * HD,
        c_enc /*cb*/, 2 * HD,
        ctx + (size_t)s * 2 * HD + HD, ctx_rs,
        bped[(t + 1) & 1], HD, nullptr, 0, HD, HD);
  }
  // final h (SRC_LEN even -> lives in ping buffer 0): h_t = [hb | hf]
  copy2d_f16_kernel<<<Bsz, 256, 0, stream>>>(hbA, HD, henc_cat,      2 * HD, HD);
  copy2d_f16_kernel<<<Bsz, 256, 0, stream>>>(hfA, HD, henc_cat + HD, 2 * HD, HD);

  // dec_h0 = tanh(h_t @ W_e2d^T + b_e2d)
  gemm(henc_cat, 2 * HD, w_e2d16, H, b_e2d, nullptr, 0, dec_h0_16, H, 1, H / 16, H, 1);

  // ---- attention decoder (sequential) ------------------------------------
  for (int t = 0; t < TRG_LEN; ++t) {
    const _Float16* hx = (t == 0) ? dec_h0_16 : (trg_h16 + (size_t)(t - 1) * Bsz * H);
    // LSTM cell: hy -> hy16 and concatA[:, 1024:2048]; c in-place (c_enc is c_t)
    lstm_step_kernel<<<H / 64, 128, 0, stream>>>(
        hx, H, w_hiddec16, dec_gx + (size_t)t * Bsz * 4 * H,
        c_enc, 2 * HD,
        nullptr, 0,
        hy16, H, concatA + H, 2 * H, H, H);
    // target = hy @ W_attn_in^T
    gemm(hy16, H, w_attnin16, H, nullptr, target32, H, nullptr, 0, 1, H / 16, H, 0);
    // attention -> wctx f16 into concatA[:, 0:1024]
    attn_kernel<<<Bsz, 256, 0, stream>>>(ctx, target32, concatA);
    // h_tilde = tanh([wctx|hy] @ W_attn_out^T) -> trg_h16[t] (also next hx)
    gemm(concatA, 2 * H, w_attnout16, 2 * H, nullptr, nullptr, 0,
         trg_h16 + (size_t)t * Bsz * H, H, 1, H / 16, 2 * H, 1);
  }

  // ---- logits = trg_h @ W_vocab^T + b_vocab ------------------------------
  gemm(trg_h16, H, w_vocab16, H, b_vocab, (float*)d_out, V, nullptr, 0,
       TRG_LEN * Bsz / 16, V / 16, H, 0);
}
